// SpanAttn_86371792323227
// MI455X (gfx1250) — matrix operand order repaired and checked by reading, not compiled, not run on hardware
//
#include <hip/hip_runtime.h>
#include <math.h>

// ---------------------------------------------------------------------------
// Problem constants (from the reference)
// ---------------------------------------------------------------------------
#define BSZ   4
#define NTOK  128
#define EMBD  768
#define BIAFD 200
#define HIDD  200
#define NH    4
#define NL    3
#define KN    10
#define NERD  9
#define HSZD  402           // 2*BIAF+2
#define SPAN  3600          // #(l,r) pairs
#define DHD   50            // HID/NH
#define HSZP  416           // HSZ padded to multiple of 32
#define HIDP  224           // HID padded to multiple of 32
#define MNALL (BSZ*NTOK*NTOK) // 65536

typedef __attribute__((ext_vector_type(16))) __bf16 v16bf;
typedef __attribute__((ext_vector_type(8)))  float  v8f;
typedef int vs4i __attribute__((vector_size(4 * sizeof(int))));

#if __has_builtin(__builtin_amdgcn_global_load_async_to_lds_b128) && \
    __has_builtin(__builtin_amdgcn_s_wait_asynccnt)
#define USE_ASYNC_LDS 1
#else
#define USE_ASYNC_LDS 0
#endif

__device__ __forceinline__ float gelu_tanh(float x) {
  float t = 0.7978845608028654f * (x + 0.044715f * x * x * x);
  return 0.5f * x * (1.0f + tanhf(t));
}

__device__ __forceinline__ float waveReduceSum(float v) {
  #pragma unroll
  for (int m = 16; m >= 1; m >>= 1) v += __shfl_xor(v, m, 32);
  return v;
}

// ---------------------------------------------------------------------------
// bf16 conversion with zero padding (A operands only)
// ---------------------------------------------------------------------------
__global__ void convert_pad_kernel(const float* __restrict__ src, __bf16* __restrict__ dst,
                                   int srcRows, int srcCols, int dstRows, int dstCols) {
  int i = blockIdx.x * blockDim.x + threadIdx.x;
  int total = dstRows * dstCols;
  if (i >= total) return;
  int r = i / dstCols, c = i % dstCols;
  float v = (r < srcRows && c < srcCols) ? src[(size_t)r * srcCols + c] : 0.0f;
  dst[i] = (__bf16)v;
}

// ---------------------------------------------------------------------------
// Pack a B matrix (f32, logically K x N) into the B-register order used here (contiguous k, as A):
//   dst[((kt*nT16 + nt16)*32 + lane)*16 + j] = B[kt*32 + (lane>>4)*16 + j][nt16*16 + (lane&15)]
// Zero-pads K up to Kp and N up to nT16*16. transpose=1: B[k][n] = src[n*srcK+k].
// ---------------------------------------------------------------------------
__global__ void pack_b_kernel(const float* __restrict__ src, __bf16* __restrict__ dst,
                              int srcK, int srcN, int Kp, int nT16tot, int transpose) {
  int i = blockIdx.x * blockDim.x + threadIdx.x;
  int total = (Kp >> 5) * nT16tot * 512;
  if (i >= total) return;
  int j    = i & 15;
  int lane = (i >> 4) & 31;
  int rem  = i >> 9;
  int nt16 = rem % nT16tot;
  int kt   = rem / nT16tot;
  int k  = kt * 32 + (lane >> 4) * 16 + j;
  int nn = nt16 * 16 + (lane & 15);
  float v = 0.0f;
  if (k < srcK && nn < srcN)
    v = transpose ? src[(size_t)nn * srcK + k] : src[(size_t)k * srcN + nn];
  dst[i] = (__bf16)v;
}

// ---------------------------------------------------------------------------
// Tiled WMMA GEMM: C[M,N] = act(A[M,K] @ B[K,N] + bias)
// Block = 128 threads = 4 waves; block tile = 64 x 64.
// - A tile (64x32 bf16) ping-pong double-buffered in LDS (stride 40: no bank
//   conflicts, 16B aligned). bf16-A staging uses GLOBAL_LOAD_ASYNC_TO_LDS_B128
//   (ASYNCcnt) when available; f32-A converts while staging.
// - B pre-swizzled into register order (pack_b_kernel), double-buffered in
//   registers; no bounds checks anywhere in the hot loop.
// - All 4 A sub-tiles are preloaded, then 4 WMMAs issue back-to-back.
// AMODE: 0 = A bf16, 1 = A f32 | EPI: 0 f32 out, 1 bf16 out, 2 scores-scatter +=
// M multiple of 64, K multiple of 32 (all shapes here qualify).
// ---------------------------------------------------------------------------
template <int AMODE, int EPI, bool GELUF>
__global__ void __launch_bounds__(128)
wmma_gemm_kernel(const void* __restrict__ Aptr, int lda,
                 const __bf16* __restrict__ Bsw, int nT16tot,
                 const float* __restrict__ bias,
                 void* __restrict__ Cout, int ldc,
                 int M, int Nc, int Kc, int nTilesN) {
  constexpr int LSTR = 40;                          // LDS row stride (bf16 elems)
  __shared__ __bf16 ldsA[2][64 * LSTR];
  int mt = blockIdx.x / nTilesN;
  int nt = blockIdx.x % nTilesN;
  int mBase = mt * 64;
  int tid   = threadIdx.x;
  int lane  = tid & 31;
  int wv    = tid >> 5;
  int khalf = lane >> 4;
  int mrow  = lane & 15;
  int n     = nt * 64 + wv * 16 + mrow;
  int nt16  = nt * 4 + wv;

  v8f acc[4];
  #pragma unroll
  for (int st = 0; st < 4; st++) {
    #pragma unroll
    for (int r = 0; r < 8; r++) acc[st][r] = 0.0f;
  }

  // Staging decomposition: thread covers 16 consecutive elems of 64x32 tile.
  int fr = tid >> 1;                                 // row 0..63
  int fc = (tid & 1) * 16;                           // col 0 or 16

  auto fillA = [&](int bufI, int kb) {
    #pragma unroll
    for (int half = 0; half < 2; half++) {
      int c = fc + half * 8;
      __bf16* dp = &ldsA[bufI][fr * LSTR + c];
      if (AMODE == 1) {
        const float* sp = (const float*)Aptr + (size_t)(mBase + fr) * lda + kb + c;
        __bf16 tmp[8];
        #pragma unroll
        for (int t = 0; t < 8; t++) tmp[t] = (__bf16)sp[t];
        *(uint4*)dp = *(const uint4*)tmp;
      } else {
        const __bf16* sp = (const __bf16*)Aptr + (size_t)(mBase + fr) * lda + kb + c;
#if USE_ASYNC_LDS
        __builtin_amdgcn_global_load_async_to_lds_b128(
            (vs4i*)(void*)sp, (vs4i*)(void*)dp, 0, 0);
#else
        *(uint4*)dp = *(const uint4*)sp;
#endif
      }
    }
  };
  auto asyncWait = [&]() {
#if USE_ASYNC_LDS
    if (AMODE == 0) __builtin_amdgcn_s_wait_asynccnt(0);
#endif
  };
  auto loadB = [&](const __bf16* p) {
    v16bf b;
    uint4* bp = (uint4*)&b;
    bp[0] = *(const uint4*)(p + 0);
    bp[1] = *(const uint4*)(p + 8);
    return b;
  };

  const __bf16* Bp = Bsw + (size_t)nt16 * 512 + (size_t)lane * 16;
  size_t bstep = (size_t)nT16tot * 512;

  // Prologue: stage tile 0, load B tile 0.
  int cur = 0;
  fillA(0, 0);
  v16bf bcur = loadB(Bp);
  Bp += bstep;
  asyncWait();
  __syncthreads();

  for (int kb = 0; kb < Kc; kb += 32) {
    bool hasNext = (kb + 32) < Kc;
    v16bf bnext;
    if (hasNext) {
      bnext = loadB(Bp);                             // overlaps with WMMAs below
      Bp += bstep;
      fillA(cur ^ 1, kb + 32);                       // fill other LDS buffer
    }
    // Speculative prefetch two tiles ahead (dropped harmlessly past the end).
    __builtin_prefetch((const char*)Aptr +
        ((size_t)(mBase + fr) * lda + kb + 64) * (AMODE == 1 ? 4 : 2), 0, 1);

    // Preload all 4 A sub-tiles, then 4 back-to-back WMMAs.
    v16bf a[4];
    #pragma unroll
    for (int st = 0; st < 4; st++) {
      // A v16bf = one contiguous 16-elem run: K = khalf*16+{0..15}
      const __bf16* ar = &ldsA[cur][(st * 16 + mrow) * LSTR + khalf * 16];
      uint4* ap = (uint4*)&a[st];
      ap[0] = *(const uint4*)(ar + 0);
      ap[1] = *(const uint4*)(ar + 8);
    }
    #pragma unroll
    for (int st = 0; st < 4; st++) {
      acc[st] = __builtin_amdgcn_wmma_f32_16x16x32_bf16(false, a[st], false, bcur,
                                                        (short)0, acc[st], false, false);
    }
    if (hasNext) {
      asyncWait();
      __syncthreads();
    }
    cur ^= 1;
    bcur = bnext;
  }

  if (n < Nc) {
    float bb = bias ? bias[n] : 0.0f;
    #pragma unroll
    for (int st = 0; st < 4; st++) {
      #pragma unroll
      for (int r = 0; r < 8; r++) {
        int m = mBase + st * 16 + r + khalf * 8;     // C/D layout: m = r + 8*khalf
        float v = acc[st][r] + bb;
        if (GELUF) v = gelu_tanh(v);
        if (EPI == 0) {
          ((float*)Cout)[(size_t)m * ldc + n] = v;
        } else if (EPI == 1) {
          ((__bf16*)Cout)[(size_t)m * ldc + n] = (__bf16)v;
        } else {                                     // scores layout (b,c,m,n), += biaf
          size_t idx = ((size_t)(m >> 14) * HIDD + n) * (size_t)(NTOK * NTOK) + (size_t)(m & 16383);
          ((float*)Cout)[idx] += v;
        }
      }
    }
  }
}

// ---------------------------------------------------------------------------
// biaf stage 1: P[b,l,h,d,y] = sum_x hh[b,l,h,x] * U[h,d,x,y]
// ---------------------------------------------------------------------------
__global__ void biaf_p_kernel(const float* __restrict__ hs, const float* __restrict__ U,
                              float* __restrict__ P) {
  int i = blockIdx.x * blockDim.x + threadIdx.x;    // (b,l,h,d)
  if (i >= BSZ * NTOK * NH * DHD) return;
  int d = i % DHD;
  int h = (i / DHD) % NH;
  int l = (i / (DHD * NH)) % NTOK;
  int b = i / (DHD * NH * NTOK);
  const float* hrow = hs + ((size_t)(b * NTOK + l)) * BIAFD + h * DHD;
  float hr[DHD];
  #pragma unroll
  for (int x = 0; x < DHD; x++) hr[x] = hrow[x];
  const float* Ub = U + (((size_t)h * DHD + d) * DHD) * DHD;
  float* Pr = P + (size_t)i * DHD;
  for (int y = 0; y < DHD; y++) {
    float accv = 0.0f;
    #pragma unroll
    for (int x = 0; x < DHD; x++) accv += hr[x] * Ub[x * DHD + y];
    Pr[y] = accv;
  }
}

// biaf stage 2: scores[b, h*50+d, l, k] = sum_y P[b,l,h,d,y] * tt[b,k,h,y]
__global__ void biaf_scores_kernel(const float* __restrict__ P, const float* __restrict__ ts,
                                   float* __restrict__ scores) {
  int i = blockIdx.x * blockDim.x + threadIdx.x;    // (b,c,l)
  if (i >= BSZ * HIDD * NTOK) return;
  int l = i % NTOK;
  int c = (i / NTOK) % HIDD;
  int b = i / (NTOK * HIDD);
  int h = c / DHD, d = c % DHD;
  const float* Pr = P + (((size_t)(b * NTOK + l) * NH + h) * DHD + d) * DHD;
  float pr[DHD];
  #pragma unroll
  for (int y = 0; y < DHD; y++) pr[y] = Pr[y];
  float* srow = scores + ((size_t)(b * HIDD + c) * NTOK + l) * NTOK;
  for (int k = 0; k < NTOK; k++) {
    const float* trow = ts + ((size_t)(b * NTOK + k)) * BIAFD + h * DHD;
    float accv = 0.0f;
    #pragma unroll
    for (int y = 0; y < DHD; y++) accv += pr[y] * trow[y];
    srow[k] = accv;
  }
}

// ---------------------------------------------------------------------------
// aff = LN(concat(h1,t1) + alpha*proj) -> bf16, stride 416 (tail zeroed)
// ---------------------------------------------------------------------------
__device__ __forceinline__ float aff_raw(int c, const float* hrow, const float* trow,
                                         const __bf16* pr, float alpha) {
  float basev;
  if (c < BIAFD)       basev = hrow[c];
  else if (c == BIAFD) basev = 1.0f;
  else {
    int cc = c - (BIAFD + 1);
    basev = (cc < BIAFD) ? trow[cc] : 1.0f;
  }
  return basev + alpha * (float)pr[c];
}

__global__ void aff_ln_kernel(const __bf16* __restrict__ proj, const float* __restrict__ hs,
                              const float* __restrict__ ts, const float* __restrict__ ln_g,
                              const float* __restrict__ ln_b, const float* __restrict__ alphap,
                              __bf16* __restrict__ aff) {
  int row  = blockIdx.x * 4 + ((int)threadIdx.x >> 5);
  int lane = threadIdx.x & 31;
  int b = row >> 14, mn = row & 16383, m = mn >> 7, n = mn & 127;
  float alpha = alphap[0];
  const __bf16* pr = proj + (size_t)row * HSZD;
  const float* hrow = hs + (size_t)(b * NTOK + m) * BIAFD;
  const float* trow = ts + (size_t)(b * NTOK + n) * BIAFD;
  float sum = 0.0f, sumsq = 0.0f;
  for (int c = lane; c < HSZD; c += 32) {
    float x = aff_raw(c, hrow, trow, pr, alpha);
    sum += x; sumsq += x * x;
  }
  sum = waveReduceSum(sum); sumsq = waveReduceSum(sumsq);
  float mean = sum / (float)HSZD;
  float var  = sumsq / (float)HSZD - mean * mean;
  float rstd = rsqrtf(var + 1e-5f);
  __bf16* arow = aff + (size_t)row * HSZP;
  for (int c = lane; c < HSZP; c += 32) {
    if (c < HSZD) {
      float x = aff_raw(c, hrow, trow, pr, alpha);
      arow[c] = (__bf16)(ln_g[c] * (x - mean) * rstd + ln_b[c]);
    } else {
      arow[c] = (__bf16)0.0f;
    }
  }
}

// ---------------------------------------------------------------------------
// feat[b,s,c] = scores[b,c,l,r]  (+ bf16 copy, stride 224, tail zeroed)
// ---------------------------------------------------------------------------
__global__ void gather_feat_kernel(const float* __restrict__ scores, const int* __restrict__ id2lr,
                                   float* __restrict__ feat, __bf16* __restrict__ featbf) {
  int i = blockIdx.x * blockDim.x + threadIdx.x;
  if (i >= BSZ * SPAN * HIDP) return;
  int c = i % HIDP;
  int s = (i / HIDP) % SPAN;
  int b = i / (HIDP * SPAN);
  size_t bs = (size_t)b * SPAN + s;
  if (c >= HIDD) { featbf[bs * HIDP + c] = (__bf16)0.0f; return; }
  int l = id2lr[bs * 2 + 0];
  int r = id2lr[bs * 2 + 1];
  float v = scores[(((size_t)(b * HIDD + c)) * NTOK + l) * NTOK + r];
  feat[bs * HIDD + c]   = v;
  featbf[bs * HIDP + c] = (__bf16)v;
}

// ---------------------------------------------------------------------------
// Attention core: one thread per (b,s,h). K/V projected once for all rows,
// gathered afterwards (projection and gather commute). Output bf16, pad 224.
// ---------------------------------------------------------------------------
__global__ void attn_core_kernel(const float* __restrict__ qbuf, const float* __restrict__ kbuf,
                                 const float* __restrict__ vbuf, const int* __restrict__ N_idx,
                                 const unsigned char* __restrict__ N_mask,
                                 const float* __restrict__ head_gate, __bf16* __restrict__ obf) {
  int gid = blockIdx.x * blockDim.x + threadIdx.x;
  if (gid >= BSZ * SPAN * NH) return;
  int h = gid & 3;
  int s = (gid >> 2) % SPAN;
  int b = gid / (SPAN * NH);
  size_t bs = (size_t)b * SPAN + s;
  const float* q = qbuf + bs * HIDD + h * DHD;
  float qv[DHD];
  #pragma unroll
  for (int j = 0; j < DHD; j++) qv[j] = q[j];
  int idxs[KN];
  float w[KN];
  const float inv_sqrt = 0.14142135623730950f;       // 1/sqrt(50)
  float mx = -1e30f;
  #pragma unroll
  for (int k = 0; k < KN; k++) {
    int idx = N_idx[bs * KN + k];
    idxs[k] = idx;
    const float* kr = kbuf + ((size_t)b * SPAN + idx) * HIDD + h * DHD;
    float d = 0.0f;
    #pragma unroll
    for (int j = 0; j < DHD; j++) d += qv[j] * kr[j];
    float lg = logf(head_gate[(((size_t)b * NH + h) * SPAN + s) * KN + k] + 1e-9f);
    float lv = N_mask[bs * KN + k] ? (d * inv_sqrt + lg) : -1e9f;
    w[k] = lv;
    mx = fmaxf(mx, lv);
  }
  float sum = 0.0f;
  #pragma unroll
  for (int k = 0; k < KN; k++) { w[k] = expf(w[k] - mx); sum += w[k]; }
  float rs = 1.0f / sum;
  float ov[DHD];
  #pragma unroll
  for (int j = 0; j < DHD; j++) ov[j] = 0.0f;
  #pragma unroll
  for (int k = 0; k < KN; k++) {
    float wk = w[k] * rs;
    const float* vr = vbuf + ((size_t)b * SPAN + idxs[k]) * HIDD + h * DHD;
    #pragma unroll
    for (int j = 0; j < DHD; j++) ov[j] += wk * vr[j];
  }
  __bf16* orow = obf + bs * HIDP;
  #pragma unroll
  for (int j = 0; j < DHD; j++) orow[h * DHD + j] = (__bf16)ov[j];
  if (h == 0) {
    for (int c = HIDD; c < HIDP; c++) orow[c] = (__bf16)0.0f;
  }
}

// feat = LN(feat + o@Wo) * rowmask ; one wave32 per row; refresh bf16 copy
__global__ void feat_ln_kernel(float* __restrict__ feat, const float* __restrict__ obuf,
                               const float* __restrict__ lng, const float* __restrict__ lnb,
                               const unsigned char* __restrict__ rowmask,
                               __bf16* __restrict__ featbf) {
  int row  = blockIdx.x * 4 + ((int)threadIdx.x >> 5);
  int lane = threadIdx.x & 31;
  float* f = feat + (size_t)row * HIDD;
  const float* o = obuf + (size_t)row * HIDD;
  float sum = 0.0f, sumsq = 0.0f;
  for (int c = lane; c < HIDD; c += 32) {
    float x = f[c] + o[c];
    sum += x; sumsq += x * x;
  }
  sum = waveReduceSum(sum); sumsq = waveReduceSum(sumsq);
  float mean = sum / (float)HIDD;
  float var  = sumsq / (float)HIDD - mean * mean;
  float rstd = rsqrtf(var + 1e-5f);
  float rm = rowmask[row] ? 1.0f : 0.0f;
  __bf16* fb = featbf + (size_t)row * HIDP;
  for (int c = lane; c < HIDP; c += 32) {
    if (c < HIDD) {
      float x = f[c] + o[c];
      float y = (lng[c] * (x - mean) * rstd + lnb[c]) * rm;
      f[c] = y;
      fb[c] = (__bf16)y;
    } else {
      fb[c] = (__bf16)0.0f;
    }
  }
}

// scores[b,c,l,r] = feat[b,s,c]
__global__ void scatter_feat_kernel(const float* __restrict__ feat, const int* __restrict__ id2lr,
                                    float* __restrict__ scores) {
  int i = blockIdx.x * blockDim.x + threadIdx.x;
  if (i >= BSZ * SPAN * HIDD) return;
  int c = i % HIDD;
  int s = (i / HIDD) % SPAN;
  int b = i / (HIDD * SPAN);
  size_t bs = (size_t)b * SPAN + s;
  int l = id2lr[bs * 2 + 0];
  int r = id2lr[bs * 2 + 1];
  scores[(((size_t)(b * HIDD + c)) * NTOK + l) * NTOK + r] = feat[bs * HIDD + c];
}

// out[b,m,n,k] = sum_c scores[b,c,m,n] * W_down[c,k] + b_down[k]
__global__ void final_kernel(const float* __restrict__ scores, const float* __restrict__ Wd,
                             const float* __restrict__ bd, float* __restrict__ out) {
  int i = blockIdx.x * blockDim.x + threadIdx.x;    // (b, mn)
  if (i >= MNALL) return;
  int mn = i & 16383;
  int b  = i >> 14;
  float accv[NERD];
  #pragma unroll
  for (int k = 0; k < NERD; k++) accv[k] = bd[k];
  for (int c = 0; c < HIDD; c++) {
    float s = scores[(((size_t)(b * HIDD + c)) << 14) + mn];
    #pragma unroll
    for (int k = 0; k < NERD; k++) accv[k] += s * Wd[c * NERD + k];
  }
  float* orow = out + ((size_t)b * 16384 + mn) * NERD;
  #pragma unroll
  for (int k = 0; k < NERD; k++) orow[k] = accv[k];
}

// ---------------------------------------------------------------------------
// Host orchestration
// ---------------------------------------------------------------------------
extern "C" void kernel_launch(void* const* d_in, const int* in_sizes, int n_in,
                              void* d_out, int out_size, void* d_ws, size_t ws_size,
                              hipStream_t stream) {
  (void)in_sizes; (void)n_in; (void)out_size; (void)ws_size;
  const float* word_rep   = (const float*)d_in[0];
  const float* soft_heads = (const float*)d_in[1];
  const float* head_gate  = (const float*)d_in[2];
  const int*   N_idx      = (const int*)d_in[3];
  const unsigned char* N_mask     = (const unsigned char*)d_in[4];
  const int*   id2lr      = (const int*)d_in[5];
  const unsigned char* S_row_mask = (const unsigned char*)d_in[6];
  const float* Wh    = (const float*)d_in[7];
  const float* bh    = (const float*)d_in[8];
  const float* Wt    = (const float*)d_in[9];
  const float* bt    = (const float*)d_in[10];
  const float* U_mhb = (const float*)d_in[11];
  const float* W_big = (const float*)d_in[12];
  const float* W_proj= (const float*)d_in[13];
  const float* ln_g  = (const float*)d_in[14];
  const float* ln_b  = (const float*)d_in[15];
  const float* alpha = (const float*)d_in[16];
  const float* Wq    = (const float*)d_in[17];
  const float* bq    = (const float*)d_in[18];
  const float* Wk    = (const float*)d_in[19];
  const float* bk    = (const float*)d_in[20];
  const float* Wv    = (const float*)d_in[21];
  const float* bv    = (const float*)d_in[22];
  const float* Wo    = (const float*)d_in[23];
  const float* bo    = (const float*)d_in[24];
  const float* lng   = (const float*)d_in[25];
  const float* lnb   = (const float*)d_in[26];
  const float* W_down= (const float*)d_in[27];
  const float* b_down= (const float*)d_in[28];
  float* out = (float*)d_out;

  char* wsb = (char*)d_ws;
  size_t off = 0;
  auto carve = [&](size_t bytes) -> void* {
    void* p = wsb + off;
    off = (off + bytes + 255) & ~(size_t)255;
    return p;
  };
  // swizzled-B sizes: (Kp/32) * nT16 * 512 bf16 elements
  const size_t WH_SW   = (size_t)(EMBD / 32) * 16 * 512;   // 768x(200->256)
  const size_t WPROJ_SW= (size_t)(EMBD / 32) * 28 * 512;   // 768x(402->448)
  const size_t WBIG_SW = (size_t)(HSZP / 32) * 16 * 512;   // 416x(200->256)
  const size_t WATT_SW = (size_t)(HIDP / 32) * 16 * 512;   // 224x(200->256), per layer

  __bf16* wr_bf    = (__bf16*)carve((size_t)BSZ * NTOK * EMBD * 2);
  __bf16* Whsw     = (__bf16*)carve(WH_SW * 2);
  __bf16* Wtsw     = (__bf16*)carve(WH_SW * 2);
  __bf16* Wprojsw  = (__bf16*)carve(WPROJ_SW * 2);
  __bf16* Wbigsw   = (__bf16*)carve(WBIG_SW * 2);
  __bf16* Wqsw     = (__bf16*)carve((size_t)NL * WATT_SW * 2);
  __bf16* Wksw     = (__bf16*)carve((size_t)NL * WATT_SW * 2);
  __bf16* Wvsw     = (__bf16*)carve((size_t)NL * WATT_SW * 2);
  __bf16* Wosw     = (__bf16*)carve((size_t)NL * WATT_SW * 2);
  float*  hsb      = (float*) carve((size_t)BSZ * NTOK * BIAFD * 4);
  float*  tsb      = (float*) carve((size_t)BSZ * NTOK * BIAFD * 4);
  float*  Pbuf     = (float*) carve((size_t)BSZ * NTOK * NH * DHD * DHD * 4);
  float*  scores   = (float*) carve((size_t)BSZ * HIDD * NTOK * NTOK * 4);
  __bf16* proj_bf  = (__bf16*)carve((size_t)MNALL * HSZD * 2);
  __bf16* aff_bf   = (__bf16*)carve((size_t)MNALL * HSZP * 2);
  float*  feat     = (float*) carve((size_t)BSZ * SPAN * HIDD * 4);
  __bf16* feat_bf  = (__bf16*)carve((size_t)BSZ * SPAN * HIDP * 2);
  float*  qbuf     = (float*) carve((size_t)BSZ * SPAN * HIDD * 4);
  float*  kbuf     = (float*) carve((size_t)BSZ * SPAN * HIDD * 4);
  float*  vbuf     = (float*) carve((size_t)BSZ * SPAN * HIDD * 4);
  float*  obuf     = (float*) carve((size_t)BSZ * SPAN * HIDD * 4);
  __bf16* o_bf     = (__bf16*)carve((size_t)BSZ * SPAN * HIDP * 2);

  auto packb = [&](const float* src, __bf16* dst, int srcK, int srcN, int Kp,
                   int nT16, int transpose) {
    int total = (Kp / 32) * nT16 * 512;
    pack_b_kernel<<<(total + 255) / 256, 256, 0, stream>>>(src, dst, srcK, srcN, Kp, nT16, transpose);
  };

  // ---- operand preparation ----
  convert_pad_kernel<<<((BSZ * NTOK * EMBD) + 255) / 256, 256, 0, stream>>>(
      word_rep, wr_bf, BSZ * NTOK, EMBD, BSZ * NTOK, EMBD);
  packb(Wh,     Whsw,    EMBD, BIAFD, EMBD, 16, 0);
  packb(Wt,     Wtsw,    EMBD, BIAFD, EMBD, 16, 0);
  packb(W_proj, Wprojsw, EMBD, HSZD,  EMBD, 28, 0);
  packb(W_big,  Wbigsw,  HSZD, HIDD,  HSZP, 16, 1);  // B[k][n] = W_big[n][k]
  for (int i = 0; i < NL; i++) {
    packb(Wq + (size_t)i * HIDD * HIDD, Wqsw + (size_t)i * WATT_SW, HIDD, HIDD, HIDP, 16, 0);
    packb(Wk + (size_t)i * HIDD * HIDD, Wksw + (size_t)i * WATT_SW, HIDD, HIDD, HIDP, 16, 0);
    packb(Wv + (size_t)i * HIDD * HIDD, Wvsw + (size_t)i * WATT_SW, HIDD, HIDD, HIDP, 16, 0);
    packb(Wo + (size_t)i * HIDD * HIDD, Wosw + (size_t)i * WATT_SW, HIDD, HIDD, HIDP, 16, 0);
  }

  // ---- hs / ts : GEMM(512x768 @ 768x200) + bias + GELU ----
  {
    int nT = (BIAFD + 63) / 64;                       // 4
    dim3 grid((BSZ * NTOK / 64) * nT), block(128);
    wmma_gemm_kernel<0, 0, true><<<grid, block, 0, stream>>>(
        (const void*)wr_bf, EMBD, Whsw, 16, bh, (void*)hsb, BIAFD,
        BSZ * NTOK, BIAFD, EMBD, nT);
    wmma_gemm_kernel<0, 0, true><<<grid, block, 0, stream>>>(
        (const void*)wr_bf, EMBD, Wtsw, 16, bt, (void*)tsb, BIAFD,
        BSZ * NTOK, BIAFD, EMBD, nT);
  }

  // ---- biaf -> scores buffer ----
  biaf_p_kernel<<<(BSZ * NTOK * NH * DHD + 127) / 128, 128, 0, stream>>>(hsb, U_mhb, Pbuf);
  biaf_scores_kernel<<<(BSZ * HIDD * NTOK + 127) / 128, 128, 0, stream>>>(Pbuf, tsb, scores);

  // ---- dominant GEMM: soft_heads (f32) @ W_proj -> proj_bf (bf16) ----
  {
    int nT = (HSZD + 63) / 64;                        // 7
    dim3 grid((MNALL / 64) * nT), block(128);
    wmma_gemm_kernel<1, 1, false><<<grid, block, 0, stream>>>(
        (const void*)soft_heads, EMBD, Wprojsw, 28, nullptr, (void*)proj_bf, HSZD,
        MNALL, HSZD, EMBD, nT);
  }

  // ---- aff = LN(concat + alpha*proj) -> bf16 (K padded to 416) ----
  aff_ln_kernel<<<MNALL / 4, 128, 0, stream>>>(proj_bf, hsb, tsb, ln_g, ln_b, alpha, aff_bf);

  // ---- scores += aff @ W_big^T  (scatter epilogue into (b,c,m,n)) ----
  {
    int nT = (HIDD + 63) / 64;                        // 4
    dim3 grid((MNALL / 64) * nT), block(128);
    wmma_gemm_kernel<0, 2, false><<<grid, block, 0, stream>>>(
        (const void*)aff_bf, HSZP, Wbigsw, 16, nullptr, (void*)scores, 0,
        MNALL, HIDD, HSZP, nT);
  }

  // ---- gather feat rows ----
  gather_feat_kernel<<<(BSZ * SPAN * HIDP + 255) / 256, 256, 0, stream>>>(scores, id2lr, feat, feat_bf);

  // ---- 3 neighbor-attention layers ----
  {
    int M = BSZ * SPAN;                               // 14400 = 225*64
    int nT = (HIDD + 63) / 64;                        // 4
    dim3 grid((M / 64) * nT), block(128);
    for (int i = 0; i < NL; i++) {
      const __bf16* wq = Wqsw + (size_t)i * WATT_SW;
      const __bf16* wk = Wksw + (size_t)i * WATT_SW;
      const __bf16* wv = Wvsw + (size_t)i * WATT_SW;
      const __bf16* wo = Wosw + (size_t)i * WATT_SW;
      wmma_gemm_kernel<0, 0, false><<<grid, block, 0, stream>>>(
          (const void*)feat_bf, HIDP, wq, 16, bq + i * HIDD, (void*)qbuf, HIDD, M, HIDD, HIDP, nT);
      wmma_gemm_kernel<0, 0, false><<<grid, block, 0, stream>>>(
          (const void*)feat_bf, HIDP, wk, 16, bk + i * HIDD, (void*)kbuf, HIDD, M, HIDD, HIDP, nT);
      wmma_gemm_kernel<0, 0, false><<<grid, block, 0, stream>>>(
          (const void*)feat_bf, HIDP, wv, 16, bv + i * HIDD, (void*)vbuf, HIDD, M, HIDD, HIDP, nT);
      attn_core_kernel<<<(BSZ * SPAN * NH + 255) / 256, 256, 0, stream>>>(
          qbuf, kbuf, vbuf, N_idx, N_mask, head_gate, o_bf);
      wmma_gemm_kernel<0, 0, false><<<grid, block, 0, stream>>>(
          (const void*)o_bf, HIDP, wo, 16, bo + i * HIDD, (void*)obuf, HIDD, M, HIDD, HIDP, nT);
      feat_ln_kernel<<<M / 4, 128, 0, stream>>>(
          feat, obuf, lng + i * HIDD, lnb + i * HIDD, S_row_mask, feat_bf);
    }
  }

  // ---- scatter feat back, final 200->9 projection ----
  scatter_feat_kernel<<<(BSZ * SPAN * HIDD + 255) / 256, 256, 0, stream>>>(feat, id2lr, scores);
  final_kernel<<<(MNALL + 255) / 256, 256, 0, stream>>>(scores, W_down, b_down, out);
}
